// TemporalAttention_54150947668425
// MI455X (gfx1250) — matrix operand order repaired and checked by reading, compiled, not run on hardware
//
#include <hip/hip_runtime.h>

// ---------------- CDNA5 WMMA types ----------------
typedef __attribute__((ext_vector_type(16))) __bf16 v16bf;
typedef __attribute__((ext_vector_type(8)))  float  v8f;
typedef __attribute__((ext_vector_type(8)))  unsigned short us8;

#define C_    256
#define TC    768          // 3*C
#define HD    64           // head dim
#define NHEAD 4
#define NT_QKV 48          // 768/16 output tiles
#define NT_PRJ 16          // 256/16 output tiles
#define KC_N   8           // 256/32 k-steps
#define TOKENS (32*4096)

union AF {
    v16bf v;
    __bf16 b[16];
    unsigned short s[16];
    us8 h[2];
};
union V8 {
    us8 u;
    __bf16 b[8];
    unsigned short s[8];
};

__device__ __forceinline__ void cvt4(__bf16* d, float4 f) {
    d[0] = (__bf16)f.x; d[1] = (__bf16)f.y;
    d[2] = (__bf16)f.z; d[3] = (__bf16)f.w;
}

// ---------------- kernel 0: pack weights to bf16 B-fragment layout ----------
// B fragment (32x16 KxN): lane=(half,n): N=lane%16, K=(e<8 ? half*8+e : 16+half*8+(e-8))
// two 8-float runs in K -> each lane writes 16 contiguous bf16 (32 B).
__global__ __launch_bounds__(256) void pack_weights(
    const float* __restrict__ wqkv, const float* __restrict__ wproj,
    unsigned short* __restrict__ pq, unsigned short* __restrict__ pp)
{
    int tid  = blockIdx.x * blockDim.x + threadIdx.x;   // 0 .. 16383
    int f    = tid >> 5;
    int lane = tid & 31;
    int l16  = lane & 15, half = lane >> 4;
    const float* src;
    unsigned short* dst;
    if (f < NT_QKV * KC_N) {
        int jn = f >> 3, kc = f & 7;
        src = wqkv + (size_t)(jn * 16 + l16) * C_ + kc * 32 + half * 8;
        dst = pq + ((size_t)f * 32 + lane) * 16;
    } else if (f < (NT_QKV + NT_PRJ) * KC_N) {
        int f2 = f - NT_QKV * KC_N;
        int jn = f2 >> 3, kc = f2 & 7;
        src = wproj + (size_t)(jn * 16 + l16) * C_ + kc * 32 + half * 8;
        dst = pp + ((size_t)f2 * 32 + lane) * 16;
    } else {
        return;
    }
    const float4* s4 = (const float4*)src;
    #pragma unroll
    for (int g = 0; g < 4; ++g) {
        V8 tmp;  // write in 8-element halves to keep stores vectorizable
        cvt4(tmp.b + 0, s4[g & ~1u ? g : g]);  // placeholder (see below)
        (void)tmp;
    }
    // straightforward: convert two 8-float K runs (src+0, src+16) -> 16 contiguous bf16
    V8 lo, hi;
    cvt4(lo.b + 0, s4[0]);
    cvt4(lo.b + 4, s4[1]);
    cvt4(hi.b + 0, s4[4]);
    cvt4(hi.b + 4, s4[5]);
    *(us8*)(dst + 0) = lo.u;
    *(us8*)(dst + 8) = hi.u;
}

// ---------------- kernel 1: fused qkv -> windowed attention -> proj ---------
#define WAVES 2
#define LDS_QKV (16 * TC * 2)          // 24576 B  bf16 [16][768]
#define LDS_OUT (16 * C_ * 2)          //  8192 B  bf16 [16][256]
#define LDS_SCO (NHEAD * 16 * 16 * 2)  //  2048 B  bf16 [4][16][16]
#define LDS_WAVE (LDS_QKV + LDS_OUT + LDS_SCO)   // 34816 B

__global__ __launch_bounds__(WAVES * 32) void attn_fused(
    const float* __restrict__ x,
    const float* __restrict__ b_qkv,
    const float* __restrict__ b_proj,
    const unsigned short* __restrict__ wq,
    const unsigned short* __restrict__ wp,
    float* __restrict__ out)
{
    extern __shared__ char smem[];
    const int lane = threadIdx.x & 31;
    const int wave = threadIdx.x >> 5;
    const int l16  = lane & 15, half = lane >> 4;
    const int tile = blockIdx.x * WAVES + wave;   // 16-token tile (2 windows)
    const size_t tok0 = (size_t)tile * 16;

    char* base = smem + (size_t)wave * LDS_WAVE;
    unsigned short* qkvs = (unsigned short*)base;                  // bf16 [16][768]
    unsigned short* outs = (unsigned short*)(base + LDS_QKV);      // bf16 [16][256]
    unsigned short* sco  = (unsigned short*)(base + LDS_QKV + LDS_OUT); // bf16 [4][16][16]

    // ---- stage 1: load x rows, convert to bf16 A-fragments --------------
    // A layout (16-bit, ISA 7.12.2): lane holds row M=l16; element e maps to
    // K = kc*32 + (e<8 ? half*8+e : 16+half*8+(e-8)) -> two 8-float runs.
    AF a[KC_N];
    const float* xrow = x + (tok0 + l16) * C_;
    #pragma unroll
    for (int kc = 0; kc < KC_N; ++kc) {
        const float4* p0 = (const float4*)(xrow + kc * 32 + half * 8);
        const float4* p1 = (const float4*)(xrow + kc * 32 + 16 + half * 8);
        cvt4(a[kc].b + 0,  p0[0]);
        cvt4(a[kc].b + 4,  p0[1]);
        cvt4(a[kc].b + 8,  p1[0]);
        cvt4(a[kc].b + 12, p1[1]);
    }

    // ---- stage 2: qkv = x @ w_qkv^T + b_qkv  (48 tiles x 8 WMMA) --------
    for (int jn = 0; jn < NT_QKV; ++jn) {
        v8f acc = {0.f, 0.f, 0.f, 0.f, 0.f, 0.f, 0.f, 0.f};
        #pragma unroll
        for (int kc = 0; kc < KC_N; ++kc) {
            const v16bf* bp = (const v16bf*)(wq + ((size_t)(jn * 8 + kc) * 32 + lane) * 16);
            v16bf bv = *bp;
            acc = __builtin_amdgcn_wmma_f32_16x16x32_bf16(
                false, a[kc].v, false, bv, (short)0, acc, false, false);
        }
        float bias = b_qkv[jn * 16 + l16];
        // D layout: lane -> (M = r + 8*half, N = l16); scatter rows to LDS
        #pragma unroll
        for (int r = 0; r < 8; ++r) {
            __bf16 q = (__bf16)(acc[r] + bias);
            qkvs[(r + 8 * half) * TC + jn * 16 + l16] = *(unsigned short*)&q;
        }
    }

    // ---- stage 3: scores = softmax(mask(K @ Q^T * scale)) ---------------
    // A = K (M = key token), B = Q^T (N = query token). For query t the 8
    // in-window key scores land in lane (t%16 + 16*(t/8))'s 8 acc VGPRs.
    const float scale = 0.125f;   // 1/sqrt(64)
    for (int h = 0; h < NHEAD; ++h) {
        AF ak0, ak1, bq0, bq1;
        const unsigned short* krow = qkvs + l16 * TC + C_ + h * HD;  // K
        const unsigned short* qrow = qkvs + l16 * TC + h * HD;       // Q
        ak0.h[0] = *(const us8*)(krow + half * 8);
        ak0.h[1] = *(const us8*)(krow + 16 + half * 8);
        ak1.h[0] = *(const us8*)(krow + 32 + half * 8);
        ak1.h[1] = *(const us8*)(krow + 48 + half * 8);
        bq0.h[0] = *(const us8*)(qrow + half * 8);
        bq0.h[1] = *(const us8*)(qrow + 16 + half * 8);
        bq1.h[0] = *(const us8*)(qrow + 32 + half * 8);
        bq1.h[1] = *(const us8*)(qrow + 48 + half * 8);

        v8f s = {0.f, 0.f, 0.f, 0.f, 0.f, 0.f, 0.f, 0.f};
        s = __builtin_amdgcn_wmma_f32_16x16x32_bf16(false, ak0.v, false, bq0.v, (short)0, s, false, false);
        s = __builtin_amdgcn_wmma_f32_16x16x32_bf16(false, ak1.v, false, bq1.v, (short)0, s, false, false);

        int  qq    = l16 & 7;                 // query index within window
        bool valid = (half == (l16 >> 3));    // key half matches query window
        float sc[8];
        float m = -1e30f;
        #pragma unroll
        for (int r = 0; r < 8; ++r) {
            sc[r] = s[r] * scale;
            if (r <= qq && sc[r] > m) m = sc[r];
        }
        float sum = 0.f;
        #pragma unroll
        for (int r = 0; r < 8; ++r) {
            float ev = (valid && r <= qq) ? __expf(sc[r] - m) : 0.f;
            sc[r] = ev; sum += ev;
        }
        float inv = valid ? (1.0f / sum) : 0.0f;   // valid lanes: sum >= 1
        // att[query][key] in bf16; 8 contiguous keys -> one b128 store
        V8 st;
        #pragma unroll
        for (int r = 0; r < 8; ++r) st.b[r] = (__bf16)(sc[r] * inv);
        *(us8*)(sco + h * 256 + l16 * 16 + 8 * half) = st.u;
    }

    // ---- stage 4: out = att @ V (K padded 16->32 with zeros) ------------
    for (int h = 0; h < NHEAD; ++h) {
        // A-fragment: e<8 -> K = half*8+e (real keys), e>=8 -> K>=16 (zero pad)
        AF aa;
        aa.h[0] = *(const us8*)(sco + h * 256 + l16 * 16 + half * 8);
        #pragma unroll
        for (int e = 8; e < 16; ++e) aa.s[e] = 0;

        #pragma unroll
        for (int jd = 0; jd < 4; ++jd) {
            // B-fragment: e<8 -> K = half*8+e (token), e>=8 -> zero pad
            AF bv;
            {
                #pragma unroll
                for (int e = 0; e < 8; ++e)
                    bv.s[e] = qkvs[(half * 8 + e) * TC + 2 * C_ + h * HD + jd * 16 + l16];
                #pragma unroll
                for (int e = 8; e < 16; ++e) bv.s[e] = 0;
            }
            v8f o = {0.f, 0.f, 0.f, 0.f, 0.f, 0.f, 0.f, 0.f};
            o = __builtin_amdgcn_wmma_f32_16x16x32_bf16(
                false, aa.v, false, bv.v, (short)0, o, false, false);
            #pragma unroll
            for (int r = 0; r < 8; ++r) {
                __bf16 ob = (__bf16)o[r];
                outs[(r + 8 * half) * C_ + h * HD + jd * 16 + l16] =
                    *(unsigned short*)&ob;
            }
        }
    }

    // ---- stage 5: y = out @ w_proj^T + b_proj ---------------------------
    AF ap[KC_N];
    #pragma unroll
    for (int kc = 0; kc < KC_N; ++kc) {
        const unsigned short* orow = outs + l16 * C_ + kc * 32;
        ap[kc].h[0] = *(const us8*)(orow + half * 8);
        ap[kc].h[1] = *(const us8*)(orow + 16 + half * 8);
    }
    for (int jn = 0; jn < NT_PRJ; ++jn) {
        v8f acc = {0.f, 0.f, 0.f, 0.f, 0.f, 0.f, 0.f, 0.f};
        #pragma unroll
        for (int kc = 0; kc < KC_N; ++kc) {
            const v16bf* bp = (const v16bf*)(wp + ((size_t)(jn * 8 + kc) * 32 + lane) * 16);
            v16bf bv = *bp;
            acc = __builtin_amdgcn_wmma_f32_16x16x32_bf16(
                false, ap[kc].v, false, bv, (short)0, acc, false, false);
        }
        float bias = b_proj[jn * 16 + l16];
        #pragma unroll
        for (int r = 0; r < 8; ++r)
            out[(tok0 + r + 8 * half) * C_ + jn * 16 + l16] = acc[r] + bias;
    }
}

// ---------------- host launch ----------------
extern "C" void kernel_launch(void* const* d_in, const int* in_sizes, int n_in,
                              void* d_out, int out_size, void* d_ws, size_t ws_size,
                              hipStream_t stream) {
    const float* x      = (const float*)d_in[0];
    const float* w_qkv  = (const float*)d_in[1];
    const float* b_qkv  = (const float*)d_in[2];
    const float* w_proj = (const float*)d_in[3];
    const float* b_proj = (const float*)d_in[4];
    float* out = (float*)d_out;

    unsigned short* pq = (unsigned short*)d_ws;                  // 384 frags * 512 elems
    unsigned short* pp = pq + (size_t)NT_QKV * KC_N * 32 * 16;   // +196608 elems

    // pack weights: (48+16)*8 fragments * 32 lanes = 16384 threads
    pack_weights<<<64, 256, 0, stream>>>(w_qkv, w_proj, pq, pp);

    const int tiles  = TOKENS / 16;          // 8192 waves of 16 tokens
    const int blocks = tiles / WAVES;        // 4096
    const size_t lds = (size_t)WAVES * LDS_WAVE;   // 69632 B
    attn_fused<<<blocks, WAVES * 32, lds, stream>>>(x, b_qkv, b_proj, pq, pp, out);
}
